// ADMM_RED_UNFOLD_27367531610605
// MI455X (gfx1250) — compile-verified, matrix operand order checked by reading, not run on hardware
//
#include <hip/hip_runtime.h>
#include <hip/hip_bf16.h>
#include <math.h>

// ---------------------------------------------------------------------------
// Reformer-style LSH attention for MI455X (gfx1250, wave32, WMMA).
// Pipeline:
//   K1 conv3x3  -> x_embed[L][8]
//   K2 conv1x1  -> y_embed[L][32]
//   K3 hashing  -> codes[4L] in [0,512)
//   K4 histogram (512 bins), K5 deterministic stable counting sort
//   K6 per-chunk attention with v_wmma_f32_16x16x32_f16 (QK^T K=8 padded to 32,
//      PV K=432 padded to 448), flash-style 2-pass softmax, all tiles in LDS
//   K7 un-sort + softmax-over-hashes merge + residual add
// ---------------------------------------------------------------------------

#define L_TOT   36864      // 192*192
#define HDIM    192
#define NH      4          // N_HASHES
#define NCHUNK  256        // L / CHUNK
#define CHUNKSZ 144
#define KPAD    32         // K=8 padded to 32 for f16 WMMA
#define JTOT    432        // 3*CHUNK keys (self + prev + next chunk)
#define JPAD    448        // padded to multiple of 32 for PV k-loop
#define NBUCK   512        // hash key space (128 buckets * 4 hashes)

typedef __attribute__((ext_vector_type(16))) _Float16 v16h;
typedef __attribute__((ext_vector_type(8)))  float    v8f;

// Per-lane K index for the 16-bit A/B fragment layout (ISA 7.12.2):
// element e of the v16h = VGPR e/2, half e%2.
// lanes 0-15 cover K in {0..7,16..23}; lanes 16-31 cover K in {8..15,24..31}.
__device__ __forceinline__ int kmap(int e, int lane) {
  int v = e >> 1, h = e & 1;
  int base = (v < 4) ? (2 * v) : (16 + 2 * (v - 4));
  return base + ((lane & 16) ? 8 : 0) + h;
}

// --------------------------- K1: 3x3 conv (C=8) ----------------------------
__global__ void conv3x3_kernel(const float* __restrict__ x,
                               const float* __restrict__ w,
                               const float* __restrict__ b,
                               float* __restrict__ xe) {
  int t = blockIdx.x * blockDim.x + threadIdx.x;   // t = l*8 + co
  if (t >= L_TOT * 8) return;
  int co = t & 7, l = t >> 3;
  int py = l / HDIM, px = l % HDIM;
  float acc = b[co];
  for (int ci = 0; ci < 32; ++ci) {
    for (int dy = 0; dy < 3; ++dy) {
      int yy = py + dy - 1;
      if (yy < 0 || yy >= HDIM) continue;
      for (int dx = 0; dx < 3; ++dx) {
        int xx = px + dx - 1;
        if (xx < 0 || xx >= HDIM) continue;
        acc += x[(ci * HDIM + yy) * HDIM + xx] * w[((co * 32 + ci) * 3 + dy) * 3 + dx];
      }
    }
  }
  xe[l * 8 + co] = acc;
}

// --------------------------- K2: 1x1 conv (C=32) ---------------------------
__global__ void conv1x1_kernel(const float* __restrict__ x,
                               const float* __restrict__ w,
                               const float* __restrict__ b,
                               float* __restrict__ ye) {
  int t = blockIdx.x * blockDim.x + threadIdx.x;   // t = l*32 + co
  if (t >= L_TOT * 32) return;
  int co = t & 31, l = t >> 5;
  float acc = b[co];
  for (int ci = 0; ci < 32; ++ci) acc += x[ci * L_TOT + l] * w[co * 32 + ci];
  ye[l * 32 + co] = acc;
}

// --------------------------- K3: LSH hashing -------------------------------
__global__ void hash_kernel(const float* __restrict__ xe,
                            const float* __restrict__ rot,   // [8][4][64]
                            int* __restrict__ codes) {
  int l = blockIdx.x * blockDim.x + threadIdx.x;
  if (l >= L_TOT) return;
  float q[8];
  for (int f = 0; f < 8; ++f) q[f] = xe[l * 8 + f];
  for (int h = 0; h < NH; ++h) {
    float best = -3.0e38f; int arg = 0;
    for (int i = 0; i < 64; ++i) {
      float r = 0.f;
      for (int f = 0; f < 8; ++f) r += q[f] * rot[(f * NH + h) * 64 + i];
      if (r > best)  { best = r;  arg = i; }
      if (-r > best) { best = -r; arg = 64 + i; }
    }
    codes[h * L_TOT + l] = arg + h * 128;
  }
}

// --------------------- K4: histogram (deterministic counts) ----------------
__global__ void zero_counts_kernel(int* __restrict__ counts) {
  int t = blockIdx.x * blockDim.x + threadIdx.x;
  if (t < NBUCK) counts[t] = 0;
}
__global__ void hist_kernel(const int* __restrict__ codes, int* __restrict__ counts) {
  int t = blockIdx.x * blockDim.x + threadIdx.x;
  if (t < NH * L_TOT) atomicAdd(&counts[codes[t]], 1);
}

// ------------------ K5: stable counting sort (deterministic) ---------------
// One block, 512 threads; thread k owns bucket k and scans all keys in order,
// reproducing a stable argsort exactly. Also builds the inverse permutation.
__global__ __launch_bounds__(NBUCK) void sort_kernel(const int* __restrict__ codes,
                                                     const int* __restrict__ counts,
                                                     int* __restrict__ indices,
                                                     int* __restrict__ undo) {
  __shared__ int offs[NBUCK];
  int t = threadIdx.x;
  offs[t] = counts[t];
  __syncthreads();
  if (t == 0) {
    int run = 0;
    for (int k = 0; k < NBUCK; ++k) { int c = offs[k]; offs[k] = run; run += c; }
  }
  __syncthreads();
  int pos = offs[t];
  for (int i = 0; i < NH * L_TOT; ++i) {
    if (codes[i] == t) { indices[pos] = i; undo[i] = pos; ++pos; }
  }
}

// ------------------- K6: per-chunk WMMA attention --------------------------
// One block per (hash, chunk). 288 threads = 9 waves; wave w owns M-tile w
// (16 of the 144 query rows). Everything staged in LDS (~192 KB dynamic).
__global__ __launch_bounds__(288) void attn_kernel(const float* __restrict__ xe,
                                                   const float* __restrict__ ye,
                                                   const int* __restrict__ indices,
                                                   float* __restrict__ ret_sorted,
                                                   float* __restrict__ bs_sorted) {
  extern __shared__ char smem[];
  _Float16* sQ = (_Float16*)smem;                      // [144][32] unnormalized
  _Float16* sK = sQ + CHUNKSZ * KPAD;                  // [432][32] normalized
  _Float16* sV = sK + JTOT * KPAD;                     // [448][32] (rows 432.. zero)
  _Float16* sS = sV + JPAD * KPAD;                     // [9][16][448] scores
  int*    sPix = (int*)(sS + 9 * 16 * JPAD);           // [432]

  const int blk = blockIdx.x;                // 0..1023
  const int h   = blk >> 8;                  // hash
  const int ck  = blk & 255;                 // chunk within hash
  const int tid = threadIdx.x;
  const int base_p = h * L_TOT;              // sorted-order base for this hash

  // Which pixel feeds each of the 432 key rows (self, prev, next chunk)
  for (int j = tid; j < JTOT; j += 288) {
    int c, r;
    if (j < CHUNKSZ)            { c = ck;                       r = j; }
    else if (j < 2 * CHUNKSZ)   { c = (ck + NCHUNK - 1) & 255;  r = j - CHUNKSZ; }
    else                        { c = (ck + 1) & 255;           r = j - 2 * CHUNKSZ; }
    sPix[j] = indices[base_p + c * CHUNKSZ + r] % L_TOT;
  }
  __syncthreads();

  // Stage Q (raw) and K (row-normalized) as f16, zero-padded to K=32
  for (int j = tid; j < JTOT; j += 288) {
    int pix = sPix[j];
    float v[8]; float ss = 0.f;
    for (int e = 0; e < 8; ++e) { v[e] = xe[pix * 8 + e]; ss += v[e] * v[e]; }
    float inv = 1.0f / fmaxf(sqrtf(ss), 5e-05f);
    for (int e = 0; e < 8; ++e) {
      sK[j * KPAD + e] = (_Float16)(v[e] * inv);
      if (j < CHUNKSZ) sQ[j * KPAD + e] = (_Float16)v[e];
    }
    for (int e = 8; e < KPAD; ++e) {
      sK[j * KPAD + e] = (_Float16)0.f;
      if (j < CHUNKSZ) sQ[j * KPAD + e] = (_Float16)0.f;
    }
  }
  // Stage V (432 real rows + 16 zero pad rows)
  for (int idx = tid; idx < JPAD * 32; idx += 288) {
    int j = idx >> 5, c = idx & 31;
    float val = (j < JTOT) ? ye[sPix[j] * 32 + c] : 0.f;
    sV[j * 32 + c] = (_Float16)val;
  }
  __syncthreads();

  const int wave  = tid >> 5;     // 0..8 -> M-tile
  const int lane  = tid & 31;
  const int m0    = wave * 16;
  const int lrow  = lane & 15;
  const int khalf = lane >> 4;
  _Float16* mS = sS + wave * 16 * JPAD;

  // A fragment: Q tile (16x32, rows m0..m0+15)
  v16h aQ;
  #pragma unroll
  for (int e = 0; e < 16; ++e) aQ[e] = sQ[(m0 + lrow) * KPAD + kmap(e, lane)];

  // ---- pass 1: row max over all 432 key columns ----
  float rmax[8];
  #pragma unroll
  for (int r = 0; r < 8; ++r) rmax[r] = -3.0e38f;
  for (int jt = 0; jt < 27; ++jt) {
    v16h b;
    #pragma unroll
    for (int e = 0; e < 16; ++e) b[e] = sK[(jt * 16 + lrow) * KPAD + kmap(e, lane)];
    v8f c = {};
    c = __builtin_amdgcn_wmma_f32_16x16x32_f16(false, aQ, false, b, (short)0, c, false, false);
    #pragma unroll
    for (int r = 0; r < 8; ++r) rmax[r] = fmaxf(rmax[r], c[r]);
  }
  for (int off = 1; off < 16; off <<= 1)
    #pragma unroll
    for (int r = 0; r < 8; ++r) rmax[r] = fmaxf(rmax[r], __shfl_xor(rmax[r], off, 32));

  // zero the 16 pad columns of this wave's score tile
  for (int idx = lane; idx < 16 * 16; idx += 32) {
    int r = idx >> 4, cc = idx & 15;
    mS[r * JPAD + JTOT + cc] = (_Float16)0.f;
  }

  // ---- pass 2: exp(raw - max), row sums, score tile to LDS ----
  float rsum[8];
  #pragma unroll
  for (int r = 0; r < 8; ++r) rsum[r] = 0.f;
  for (int jt = 0; jt < 27; ++jt) {
    v16h b;
    #pragma unroll
    for (int e = 0; e < 16; ++e) b[e] = sK[(jt * 16 + lrow) * KPAD + kmap(e, lane)];
    v8f c = {};
    c = __builtin_amdgcn_wmma_f32_16x16x32_f16(false, aQ, false, b, (short)0, c, false, false);
    #pragma unroll
    for (int r = 0; r < 8; ++r) {
      float ev = __expf(c[r] - rmax[r]);
      rsum[r] += ev;
      mS[(r + 8 * khalf) * JPAD + jt * 16 + lrow] = (_Float16)ev;
    }
  }
  for (int off = 1; off < 16; off <<= 1)
    #pragma unroll
    for (int r = 0; r < 8; ++r) rsum[r] += __shfl_xor(rsum[r], off, 32);

  // bucket_score = logsumexp
  if (lrow == 0) {
    #pragma unroll
    for (int r = 0; r < 8; ++r)
      bs_sorted[base_p + ck * CHUNKSZ + m0 + r + 8 * khalf] = rmax[r] + __logf(rsum[r]);
  }
  __syncthreads();

  // ---- P @ V : 16x32 output tile, K = 448 in 14 steps of 32 ----
  v8f acc0 = {}, acc1 = {};
  for (int kt = 0; kt < 14; ++kt) {
    v16h a, b0, b1;
    #pragma unroll
    for (int e = 0; e < 16; ++e) {
      int kk = kt * 32 + kmap(e, lane);
      a[e]  = mS[lrow * JPAD + kk];
      b0[e] = sV[kk * 32 + lrow];
      b1[e] = sV[kk * 32 + 16 + lrow];
    }
    acc0 = __builtin_amdgcn_wmma_f32_16x16x32_f16(false, a, false, b0, (short)0, acc0, false, false);
    acc1 = __builtin_amdgcn_wmma_f32_16x16x32_f16(false, a, false, b1, (short)0, acc1, false, false);
  }

  // normalize by row sums and write to sorted-order output
  #pragma unroll
  for (int r = 0; r < 8; ++r) {
    int row = r + 8 * khalf;
    float inv = 1.0f / rsum[r];
    int gp = base_p + ck * CHUNKSZ + m0 + row;
    ret_sorted[gp * 32 + lrow]      = acc0[r] * inv;
    ret_sorted[gp * 32 + 16 + lrow] = acc1[r] * inv;
  }
}

// ------------------- K7: un-sort + merge hashes + residual -----------------
__global__ void combine_kernel(const int* __restrict__ undo,
                               const float* __restrict__ bs_sorted,
                               const float* __restrict__ ret_sorted,
                               const float* __restrict__ x,
                               float* __restrict__ out) {
  int l = blockIdx.x * blockDim.x + threadIdx.x;
  if (l >= L_TOT) return;
  int p[NH]; float bs[NH];
  float m = -3.0e38f;
  #pragma unroll
  for (int h = 0; h < NH; ++h) {
    p[h] = undo[h * L_TOT + l];
    bs[h] = bs_sorted[p[h]];
    m = fmaxf(m, bs[h]);
  }
  float e[NH], s = 0.f;
  #pragma unroll
  for (int h = 0; h < NH; ++h) { e[h] = __expf(bs[h] - m); s += e[h]; }
  float inv = 1.0f / s;
  for (int c = 0; c < 32; ++c) {
    float acc = 0.f;
    #pragma unroll
    for (int h = 0; h < NH; ++h) acc += e[h] * inv * ret_sorted[p[h] * 32 + c];
    out[c * L_TOT + l] = x[c * L_TOT + l] + acc;   // RES_SCALE == 1.0
  }
}

// ---------------------------------------------------------------------------
extern "C" void kernel_launch(void* const* d_in, const int* in_sizes, int n_in,
                              void* d_out, int out_size, void* d_ws, size_t ws_size,
                              hipStream_t stream) {
  const float* x       = (const float*)d_in[0];   // (1,32,192,192)
  const float* w_match = (const float*)d_in[1];   // (8,32,3,3)
  const float* b_match = (const float*)d_in[2];   // (8,)
  const float* w_asm   = (const float*)d_in[3];   // (32,32,1,1)
  const float* b_asm   = (const float*)d_in[4];   // (32,)
  const float* rot     = (const float*)d_in[5];   // (8,4,64)
  float* out = (float*)d_out;

  char* ws = (char*)d_ws;
  size_t off = 0;
  float* xe      = (float*)(ws + off); off += (size_t)L_TOT * 8 * 4;        // x_embed
  float* ye      = (float*)(ws + off); off += (size_t)L_TOT * 32 * 4;       // y_embed
  int*   codes   = (int*)  (ws + off); off += (size_t)NH * L_TOT * 4;
  int*   counts  = (int*)  (ws + off); off += (size_t)NBUCK * 4;
  int*   indices = (int*)  (ws + off); off += (size_t)NH * L_TOT * 4;
  int*   undo    = (int*)  (ws + off); off += (size_t)NH * L_TOT * 4;
  float* bss     = (float*)(ws + off); off += (size_t)NH * L_TOT * 4;
  float* rets    = (float*)(ws + off); off += (size_t)NH * L_TOT * 32 * 4;
  // total ~27.1 MB of workspace

  conv3x3_kernel<<<(L_TOT * 8 + 255) / 256, 256, 0, stream>>>(x, w_match, b_match, xe);
  conv1x1_kernel<<<(L_TOT * 32 + 255) / 256, 256, 0, stream>>>(x, w_asm, b_asm, ye);
  hash_kernel<<<(L_TOT + 255) / 256, 256, 0, stream>>>(xe, rot, codes);
  zero_counts_kernel<<<2, 256, 0, stream>>>(counts);
  hist_kernel<<<(NH * L_TOT + 255) / 256, 256, 0, stream>>>(codes, counts);
  sort_kernel<<<1, NBUCK, 0, stream>>>(codes, counts, indices, undo);

  size_t shmem = (size_t)(CHUNKSZ * KPAD + JTOT * KPAD + JPAD * KPAD + 9 * 16 * JPAD)
                 * sizeof(_Float16) + (size_t)JTOT * sizeof(int);   // ~196 KB < 320 KB/WGP
  attn_kernel<<<NH * NCHUNK, 288, shmem, stream>>>(xe, ye, indices, rets, bss);

  combine_kernel<<<(L_TOT + 255) / 256, 256, 0, stream>>>(undo, bss, rets, x, out);
}
